// SelfAttention2d_21698174779806
// MI455X (gfx1250) — matrix operand order repaired and checked by reading, not compiled, not run on hardware
//
#include <hip/hip_runtime.h>
#include <hip/hip_bf16.h>

typedef _Float16 h16;
typedef __attribute__((ext_vector_type(16))) _Float16 v16h;
typedef __attribute__((ext_vector_type(8)))  _Float16 v8h;
typedef __attribute__((ext_vector_type(8)))  float    v8f;
typedef __attribute__((ext_vector_type(4)))  int      i4_;

#define B_    32
#define C_    256
#define HW_   1024
#define NH_   4
#define DH_   64
#define CPG_  8     // channels per group (C/GROUPS = 256/32)

// ---------------------------------------------------------------------------
// Async global->LDS staging (CDNA5 GLOBAL_LOAD_ASYNC_TO_LDS_B128, ASYNCcnt).
// Builtin signature (from hipcc diagnostic): (int4 AS1*, int4 AS3*, Ii, Ii).
// Guarded: falls back to synchronous copy if the builtin is unavailable.
// ---------------------------------------------------------------------------
#if defined(__AMDGCN__) && __has_builtin(__builtin_amdgcn_global_load_async_to_lds_b128)
#define HAVE_ASYNC 1
#else
#define HAVE_ASYNC 0
#endif

#if HAVE_ASYNC
#if __has_builtin(__builtin_amdgcn_s_wait_asynccnt)
#define WAIT_ASYNC(n) __builtin_amdgcn_s_wait_asynccnt(n)
#else
#define WAIT_ASYNC(n) asm volatile("s_wait_asynccnt %0" ::"i"(n) : "memory")
#endif
#else
#define WAIT_ASYNC(n) do {} while (0)
#endif

// Copy 16 bytes global -> LDS. Async path bypasses VGPRs (tracked by ASYNCcnt).
__device__ __forceinline__ void cp_b128(const h16* g, h16* l) {
#if HAVE_ASYNC
  typedef __attribute__((address_space(1))) i4_ gi4;
  typedef __attribute__((address_space(3))) i4_ li4;
  gi4* gp = (gi4*)(uintptr_t)g;
  li4* lp = (li4*)(uint32_t)(uintptr_t)l;
  __builtin_amdgcn_global_load_async_to_lds_b128(gp, lp, 0, 0);
#else
  *(v8h*)l = *(const v8h*)g;
#endif
}

// ---------------------------------------------------------------------------
// WMMA helpers (CDNA5 V_WMMA_F32_16X16X32_F16, wave32)
// ---------------------------------------------------------------------------
__device__ __forceinline__ v8f wmma_f16(v16h a, v16h b, v8f c) {
  return __builtin_amdgcn_wmma_f32_16x16x32_f16(false, a, false, b, (short)0, c,
                                                false, false);
}

// A-matrix 16x32 f16 fragment (ISA 7.12.2): lane L<16 holds M=L, K in
// {0..7,16..23}; lane L>=16 holds M=L-16, K in {8..15,24..31}.
__device__ __forceinline__ v16h load_a_frag(const h16* p0, int ld, int lane) {
  const h16* p = p0 + (size_t)(lane & 15) * ld + ((lane >> 4) << 3);
  v8h lo = *(const v8h*)(p);
  v8h hi = *(const v8h*)(p + 16);
  v16h a;
#pragma unroll
  for (int t = 0; t < 8; ++t) { a[t] = lo[t]; a[t + 8] = hi[t]; }
  return a;
}

// B-matrix 32x16 f16 fragment: lane L<16 holds column N=L, K in {0..7,16..23};
// lane L>=16 holds N=L-16, K in {8..15,24..31}. Source is [N][K] with column stride `ld`.
__device__ __forceinline__ v16h load_b_frag(const h16* p0, int ld, int lane) {
  const h16* p = p0 + (size_t)(lane & 15) * ld + ((lane >> 4) << 3);
  v8h lo = *(const v8h*)(p);
  v8h hi = *(const v8h*)(p + 16);
  v16h b;
#pragma unroll
  for (int t = 0; t < 8; ++t) { b[t] = lo[t]; b[t + 8] = hi[t]; }
  return b;
}

// ---------------------------------------------------------------------------
// fp32 -> f16 weight conversion
// ---------------------------------------------------------------------------
__global__ __launch_bounds__(256) void cvt_kernel(const float* __restrict__ a,
                                                  h16* __restrict__ o, int n) {
  int i = blockIdx.x * 256 + threadIdx.x;
  if (i < n) o[i] = (h16)a[i];
}

// ---------------------------------------------------------------------------
// GroupNorm: block per (batch, group). Writes xnT[b][i][c] (f16, K-major).
// ---------------------------------------------------------------------------
__global__ __launch_bounds__(256) void gn_kernel(const float* __restrict__ x,
                                                 const float* __restrict__ gw,
                                                 const float* __restrict__ gb,
                                                 h16* __restrict__ xnT) {
  int b  = blockIdx.x >> 5;
  int g  = blockIdx.x & 31;
  int c0 = g * CPG_;
  const float* xp = x + ((size_t)b * C_ + c0) * HW_;

  float s = 0.f, s2 = 0.f;
  for (int idx = threadIdx.x; idx < CPG_ * HW_; idx += 256) {
    float v = xp[idx];
    s += v; s2 += v * v;
  }
  __shared__ float rs[256], rs2[256];
  rs[threadIdx.x] = s; rs2[threadIdx.x] = s2;
  __syncthreads();
  for (int st = 128; st > 0; st >>= 1) {
    if (threadIdx.x < st) {
      rs[threadIdx.x]  += rs[threadIdx.x + st];
      rs2[threadIdx.x] += rs2[threadIdx.x + st];
    }
    __syncthreads();
  }
  const float inv_n = 1.f / (float)(CPG_ * HW_);
  float mean = rs[0] * inv_n;
  float var  = rs2[0] * inv_n - mean * mean;
  float rstd = rsqrtf(var + 1e-5f);

  float wv[CPG_], bv[CPG_];
#pragma unroll
  for (int cl = 0; cl < CPG_; ++cl) {
    wv[cl] = gw[c0 + cl] * rstd;
    bv[cl] = gb[c0 + cl] - mean * wv[cl];
  }
  for (int i = threadIdx.x; i < HW_; i += 256) {
    v8h o;
#pragma unroll
    for (int cl = 0; cl < CPG_; ++cl)
      o[cl] = (h16)(xp[(size_t)cl * HW_ + i] * wv[cl] + bv[cl]);
    *(v8h*)&xnT[((size_t)b * HW_ + i) * C_ + c0] = o;  // 16B packed store
  }
}

// ---------------------------------------------------------------------------
// QKV GEMM: qkv[o,i] = sum_c W[o,c]*xnT[i,c] + bias. 128x128 tile per block,
// double-buffered 64-wide K slabs staged with async global->LDS copies.
// 8 waves in 4(o) x 2(i); each wave 32x64 -> 2x4 WMMA accumulators.
// ---------------------------------------------------------------------------
__global__ __launch_bounds__(256) void qkv_gemm_kernel(
    const h16* __restrict__ W, const float* __restrict__ bias,
    const h16* __restrict__ xnT, h16* __restrict__ q, h16* __restrict__ k,
    h16* __restrict__ vT) {
  int i0 = blockIdx.x * 128;
  int o0 = blockIdx.y * 128;
  int b  = blockIdx.z;
  int tid = threadIdx.x, wave = tid >> 5, lane = tid & 31;
  int wr = wave & 3, wc = wave >> 2;

  __shared__ __align__(64) h16 lW[2][128][64];  // [buf][o][c]  row-major (A)
  __shared__ __align__(64) h16 lX[2][128][64];  // [buf][i][c]  K-major   (B)

  // staging: 256 threads, each owns 64B of W and 64B of X per slab
  int srow = tid >> 1, spart = (tid & 1) * 32;
  const h16* gW = &W[(size_t)(o0 + srow) * C_ + spart];
  const h16* gX = &xnT[((size_t)b * HW_ + i0 + srow) * C_ + spart];

  v8f acc[2][4];
#pragma unroll
  for (int a = 0; a < 2; ++a)
#pragma unroll
    for (int j = 0; j < 4; ++j) acc[a][j] = (v8f)0.f;

#define QKV_STAGE(slab)                                                     \
  do {                                                                      \
    int bufi_ = (slab) & 1;                                                 \
    _Pragma("unroll") for (int t = 0; t < 4; ++t) {                         \
      cp_b128(gW + (slab) * 64 + t * 8, &lW[bufi_][srow][spart + t * 8]);   \
      cp_b128(gX + (slab) * 64 + t * 8, &lX[bufi_][srow][spart + t * 8]);   \
    }                                                                       \
  } while (0)

  QKV_STAGE(0);
  for (int s = 0; s < 4; ++s) {
    if (s < 3) {
      QKV_STAGE(s + 1);
      WAIT_ASYNC(8);   // 8 next-slab ops in flight => current slab complete
    } else {
      WAIT_ASYNC(0);
    }
    __syncthreads();
    const h16 (*cW)[64] = lW[s & 1];
    const h16 (*cX)[64] = lX[s & 1];
#pragma unroll
    for (int kk = 0; kk < 64; kk += 32) {
      v16h a0 = load_a_frag(&cW[wr * 32][kk], 64, lane);
      v16h a1 = load_a_frag(&cW[wr * 32 + 16][kk], 64, lane);
#pragma unroll
      for (int fj = 0; fj < 4; ++fj) {
        v16h bb = load_b_frag(&cX[wc * 64 + fj * 16][kk], 64, lane);
        acc[0][fj] = wmma_f16(a0, bb, acc[0][fj]);
        acc[1][fj] = wmma_f16(a1, bb, acc[1][fj]);
      }
    }
    __syncthreads();  // all reads done before next slab overwrites this buf
  }
#undef QKV_STAGE

  const float qscale = 0.125f;  // d^-0.5, d = 64
#pragma unroll
  for (int fr = 0; fr < 2; ++fr)
#pragma unroll
    for (int fj = 0; fj < 4; ++fj) {
      int ob = o0 + wr * 32 + fr * 16 + ((lane >> 4) << 3);  // 8 contiguous o
      int ii = i0 + wc * 64 + fj * 16 + (lane & 15);
      if (ob < C_) {                       // Q (scaled)
        int n = ob >> 6, dd = ob & 63;
        v8h pk;
#pragma unroll
        for (int r = 0; r < 8; ++r)
          pk[r] = (h16)((acc[fr][fj][r] + bias[ob + r]) * qscale);
        *(v8h*)&q[(((size_t)b * NH_ + n) * HW_ + ii) * DH_ + dd] = pk;
      } else if (ob < 2 * C_) {            // K
        int oo = ob - C_;
        int n = oo >> 6, dd = oo & 63;
        v8h pk;
#pragma unroll
        for (int r = 0; r < 8; ++r)
          pk[r] = (h16)(acc[fr][fj][r] + bias[ob + r]);
        *(v8h*)&k[(((size_t)b * NH_ + n) * HW_ + ii) * DH_ + dd] = pk;
      } else {                             // V transposed [dd][pos]
        int oo = ob - 2 * C_;
        int n = oo >> 6, dd = oo & 63;
#pragma unroll
        for (int r = 0; r < 8; ++r)
          vT[(((size_t)b * NH_ + n) * DH_ + dd + r) * HW_ + ii] =
              (h16)(acc[fr][fj][r] + bias[ob + r]);
      }
    }
}

// ---------------------------------------------------------------------------
// Flash attention: block per (i-tile=128, head, batch); wave owns 16 rows.
// K/V tiles double-buffered via async global->LDS; S = Q Kᵀ (WMMA), online
// softmax (shfl_xor over the 16-lane half that holds a row), P staged via
// per-wave LDS tile (D-layout -> A-layout), O += P V (WMMA).
// ---------------------------------------------------------------------------
__global__ __launch_bounds__(256) void attn_kernel(const h16* __restrict__ q,
                                                   const h16* __restrict__ k,
                                                   const h16* __restrict__ vT,
                                                   h16* __restrict__ attnT) {
  int n = blockIdx.y, b = blockIdx.z;
  int tid = threadIdx.x, wave = tid >> 5, lane = tid & 31;
  int i0 = blockIdx.x * 128 + wave * 16;

  const h16* qbase = q + (((size_t)b * NH_ + n) * HW_ + i0) * DH_;
  v16h qa0 = load_a_frag(qbase, DH_, lane);        // K(dd) 0..31
  v16h qa1 = load_a_frag(qbase + 32, DH_, lane);   // K(dd) 32..63

  __shared__ __align__(64) h16 lK[2][64][64];      // [buf][j][dd]
  __shared__ __align__(64) h16 lV[2][64][64];      // [buf][dd][j]
  __shared__ __align__(64) h16 lP[8][16][64];      // per-wave P tile, [i][j]

  v8f acc[4];
#pragma unroll
  for (int f = 0; f < 4; ++f) acc[f] = (v8f)0.f;
  float rowmax[8], rowsum[8];
#pragma unroll
  for (int r = 0; r < 8; ++r) { rowmax[r] = -3.0e38f; rowsum[r] = 0.f; }

  const h16* kg = k + ((size_t)b * NH_ + n) * HW_ * DH_;
  const h16* vg = vT + ((size_t)b * NH_ + n) * DH_ * HW_;

  // staging: 256 threads, each owns 32B of K and 32B of V per 64-wide tile
  int srow = tid >> 2, spart = (tid & 3) * 16;
  const h16* gK = kg + (size_t)srow * DH_ + spart;
  const h16* gV = vg + (size_t)srow * HW_ + spart;

#define ATTN_STAGE(jt)                                                        \
  do {                                                                        \
    int bufi_ = (jt) & 1;                                                     \
    _Pragma("unroll") for (int t = 0; t < 2; ++t) {                           \
      cp_b128(gK + (size_t)(jt) * 64 * DH_ + t * 8,                           \
              &lK[bufi_][srow][spart + t * 8]);                               \
      cp_b128(gV + (jt) * 64 + t * 8, &lV[bufi_][srow][spart + t * 8]);       \
    }                                                                         \
  } while (0)

  ATTN_STAGE(0);
  for (int jt = 0; jt < HW_ / 64; ++jt) {
    if (jt < HW_ / 64 - 1) {
      ATTN_STAGE(jt + 1);
      WAIT_ASYNC(4);   // 4 next-tile ops in flight => current tile complete
    } else {
      WAIT_ASYNC(0);
    }
    __syncthreads();
    const h16 (*cK)[64] = lK[jt & 1];
    const h16 (*cV)[64] = lV[jt & 1];

    // ---- scores S[16][64] = Q Kᵀ ----
    v8f s[4];
#pragma unroll
    for (int fj = 0; fj < 4; ++fj) {
      v8f c = (v8f)0.f;
      c = wmma_f16(qa0, load_b_frag(&cK[fj * 16][0], 64, lane), c);
      c = wmma_f16(qa1, load_b_frag(&cK[fj * 16][32], 64, lane), c);
      s[fj] = c;
    }

    // ---- online softmax ----
    float mnew[8];
#pragma unroll
    for (int r = 0; r < 8; ++r) mnew[r] = rowmax[r];
#pragma unroll
    for (int fj = 0; fj < 4; ++fj)
#pragma unroll
      for (int r = 0; r < 8; ++r) mnew[r] = fmaxf(mnew[r], s[fj][r]);
#pragma unroll
    for (int r = 0; r < 8; ++r)
#pragma unroll
      for (int mm = 1; mm < 16; mm <<= 1)
        mnew[r] = fmaxf(mnew[r], __shfl_xor(mnew[r], mm, 32));

    float alpha[8], psum[8];
#pragma unroll
    for (int r = 0; r < 8; ++r) {
      alpha[r] = __expf(rowmax[r] - mnew[r]);
      rowmax[r] = mnew[r];
      psum[r] = 0.f;
    }
#pragma unroll
    for (int fj = 0; fj < 4; ++fj)
#pragma unroll
      for (int r = 0; r < 8; ++r) {
        float e = __expf(s[fj][r] - mnew[r]);
        s[fj][r] = e;
        psum[r] += e;
      }
#pragma unroll
    for (int r = 0; r < 8; ++r) {
#pragma unroll
      for (int mm = 1; mm < 16; mm <<= 1)
        psum[r] += __shfl_xor(psum[r], mm, 32);
      rowsum[r] = rowsum[r] * alpha[r] + psum[r];
    }
#pragma unroll
    for (int f = 0; f < 4; ++f)
#pragma unroll
      for (int r = 0; r < 8; ++r) acc[f][r] *= alpha[r];

    // ---- stage P into per-wave LDS tile (D-layout -> row-major) ----
    {
      int prow = (lane >> 4) << 3, pcol = lane & 15;
#pragma unroll
      for (int fj = 0; fj < 4; ++fj)
#pragma unroll
        for (int r = 0; r < 8; ++r)
          lP[wave][prow + r][fj * 16 + pcol] = (h16)s[fj][r];
    }
    __syncthreads();

    // ---- O += P V ----
    v16h pa0 = load_a_frag(&lP[wave][0][0], 64, lane);
    v16h pa1 = load_a_frag(&lP[wave][0][32], 64, lane);
#pragma unroll
    for (int f = 0; f < 4; ++f) {
      acc[f] = wmma_f16(pa0, load_b_frag(&cV[f * 16][0], 64, lane), acc[f]);
      acc[f] = wmma_f16(pa1, load_b_frag(&cV[f * 16][32], 64, lane), acc[f]);
    }
    __syncthreads();  // buffer reuse fence (reads done before next stage)
  }
#undef ATTN_STAGE

  // normalize and store attnT[b][i][c] (K-major for the proj GEMM)
#pragma unroll
  for (int f = 0; f < 4; ++f) {
    int dd = f * 16 + (lane & 15);
#pragma unroll
    for (int r = 0; r < 8; ++r) {
      int ii = i0 + r + ((lane >> 4) << 3);
      attnT[((size_t)b * HW_ + ii) * C_ + n * DH_ + dd] =
          (h16)(acc[f][r] / rowsum[r]);
    }
  }
}

// ---------------------------------------------------------------------------
// Proj GEMM + bias + residual: out[b,o,i] = x + b[o] + sum_c Wp[o,c]*attnT[i,c]
// ---------------------------------------------------------------------------
__global__ __launch_bounds__(256) void proj_gemm_kernel(
    const h16* __restrict__ W, const float* __restrict__ bias,
    const h16* __restrict__ attnT, const float* __restrict__ x,
    float* __restrict__ out) {
  int i0 = blockIdx.x * 128;
  int o0 = blockIdx.y * 128;
  int b  = blockIdx.z;
  int tid = threadIdx.x, wave = tid >> 5, lane = tid & 31;
  int wr = wave & 3, wc = wave >> 2;

  __shared__ __align__(64) h16 lW[2][128][64];
  __shared__ __align__(64) h16 lX[2][128][64];

  int srow = tid >> 1, spart = (tid & 1) * 32;
  const h16* gW = &W[(size_t)(o0 + srow) * C_ + spart];
  const h16* gX = &attnT[((size_t)b * HW_ + i0 + srow) * C_ + spart];

  v8f acc[2][4];
#pragma unroll
  for (int a = 0; a < 2; ++a)
#pragma unroll
    for (int j = 0; j < 4; ++j) acc[a][j] = (v8f)0.f;

#define PRJ_STAGE(slab)                                                     \
  do {                                                                      \
    int bufi_ = (slab) & 1;                                                 \
    _Pragma("unroll") for (int t = 0; t < 4; ++t) {                         \
      cp_b128(gW + (slab) * 64 + t * 8, &lW[bufi_][srow][spart + t * 8]);   \
      cp_b128(gX + (slab) * 64 + t * 8, &lX[bufi_][srow][spart + t * 8]);   \
    }                                                                       \
  } while (0)

  PRJ_STAGE(0);
  for (int s = 0; s < 4; ++s) {
    if (s < 3) {
      PRJ_STAGE(s + 1);
      WAIT_ASYNC(8);
    } else {
      WAIT_ASYNC(0);
    }
    __syncthreads();
    const h16 (*cW)[64] = lW[s & 1];
    const h16 (*cX)[64] = lX[s & 1];
#pragma unroll
    for (int kk = 0; kk < 64; kk += 32) {
      v16h a0 = load_a_frag(&cW[wr * 32][kk], 64, lane);
      v16h a1 = load_a_frag(&cW[wr * 32 + 16][kk], 64, lane);
#pragma unroll
      for (int fj = 0; fj < 4; ++fj) {
        v16h bb = load_b_frag(&cX[wc * 64 + fj * 16][kk], 64, lane);
        acc[0][fj] = wmma_f16(a0, bb, acc[0][fj]);
        acc[1][fj] = wmma_f16(a1, bb, acc[1][fj]);
      }
    }
    __syncthreads();
  }
#undef PRJ_STAGE

#pragma unroll
  for (int fr = 0; fr < 2; ++fr)
#pragma unroll
    for (int fj = 0; fj < 4; ++fj) {
      int ob = o0 + wr * 32 + fr * 16 + ((lane >> 4) << 3);
      int ii = i0 + wc * 64 + fj * 16 + (lane & 15);
#pragma unroll
      for (int r = 0; r < 8; ++r) {
        size_t idx = ((size_t)b * C_ + ob + r) * HW_ + ii;
        out[idx] = x[idx] + bias[ob + r] + acc[fr][fj][r];
      }
    }
}

// ---------------------------------------------------------------------------
// Launch
// ---------------------------------------------------------------------------
extern "C" void kernel_launch(void* const* d_in, const int* in_sizes, int n_in,
                              void* d_out, int out_size, void* d_ws,
                              size_t ws_size, hipStream_t stream) {
  (void)in_sizes; (void)n_in; (void)out_size; (void)ws_size;
  const float* x      = (const float*)d_in[0];
  const float* gn_w   = (const float*)d_in[1];
  const float* gn_b   = (const float*)d_in[2];
  const float* qkv_w  = (const float*)d_in[3];
  const float* qkv_b  = (const float*)d_in[4];
  const float* proj_w = (const float*)d_in[5];
  const float* proj_b = (const float*)d_in[6];
  float* out = (float*)d_out;

  char* arena = (char*)d_ws;
  size_t off = 0;
  auto take = [&](size_t bytes) -> char* {
    char* p = arena + off;
    off += (bytes + 255) & ~(size_t)255;
    return p;
  };
  h16* wqkv  = (h16*)take((size_t)3 * C_ * C_ * sizeof(h16));
  h16* wproj = (h16*)take((size_t)C_ * C_ * sizeof(h16));
  h16* xnT   = (h16*)take((size_t)B_ * HW_ * C_ * sizeof(h16));
  h16* qb    = (h16*)take((size_t)B_ * NH_ * HW_ * DH_ * sizeof(h16));
  h16* kb    = (h16*)take((size_t)B_ * NH_ * HW_ * DH_ * sizeof(h16));
  h16* vTb   = (h16*)take((size_t)B_ * NH_ * DH_ * HW_ * sizeof(h16));
  h16* attnT = (h16*)take((size_t)B_ * HW_ * C_ * sizeof(h16));

  cvt_kernel<<<(3 * C_ * C_ + 255) / 256, 256, 0, stream>>>(qkv_w, wqkv,
                                                            3 * C_ * C_);
  cvt_kernel<<<(C_ * C_ + 255) / 256, 256, 0, stream>>>(proj_w, wproj,
                                                        C_ * C_);
  gn_kernel<<<B_ * 32, 256, 0, stream>>>(x, gn_w, gn_b, xnT);
  qkv_gemm_kernel<<<dim3(HW_ / 128, (3 * C_) / 128, B_), 256, 0, stream>>>(
      wqkv, qkv_b, xnT, qb, kb, vTb);
  attn_kernel<<<dim3(HW_ / 128, NH_, B_), 256, 0, stream>>>(qb, kb, vTb,
                                                            attnT);
  proj_gemm_kernel<<<dim3(HW_ / 128, C_ / 128, B_), 256, 0, stream>>>(
      wproj, proj_b, attnT, x, out);
}
